// VishwamAIModel_69681549410643
// MI455X (gfx1250) — compile-verified, matrix operand order checked by reading, not run on hardware
//
#include <hip/hip_runtime.h>
#include <hip/hip_bf16.h>

// ---------------------------------------------------------------------------
// Types for CDNA5 WMMA
// ---------------------------------------------------------------------------
typedef __attribute__((ext_vector_type(16))) __bf16 bf16x16;
typedef __attribute__((ext_vector_type(8)))  float  floatx8;

#define D_DIM 512
#define H_DIM 2048
#define E_DIM 8
#define BS_TOK 65536          // 16 * 4096 tokens
#define LDA 40                // padded LDS row stride in halves (80B: 16B aligned, conflict-free)

__device__ __forceinline__ bf16x16 load_frag(const __bf16* rowbase, int kb) {
    union { bf16x16 v; uint4 q[2]; } u;
    u.q[0] = *reinterpret_cast<const uint4*>(rowbase + kb);
    u.q[1] = *reinterpret_cast<const uint4*>(rowbase + kb + 16);
    return u.v;
}

__device__ __forceinline__ float silu_f(float x) {
    return x / (1.0f + __expf(-x));
}

// gfx1250 async DMA: 16B per lane, global -> LDS, tracked by ASYNCcnt.
__device__ __forceinline__ void async_b128(unsigned lds_addr, unsigned long long gaddr) {
    asm volatile("global_load_async_to_lds_b128 %0, %1, off"
                 :: "v"(lds_addr), "v"(gaddr) : "memory");
}

__device__ __forceinline__ void wait_async0() {
    asm volatile("s_wait_asynccnt 0x0" ::: "memory");
}

// ---------------------------------------------------------------------------
// Kernel 0a: zero the expert-usage accumulator (8 floats)
// ---------------------------------------------------------------------------
__global__ void k_init(float* usage) {
    if (threadIdx.x < E_DIM) usage[threadIdx.x] = 0.0f;
}

// ---------------------------------------------------------------------------
// Kernel 0b: transpose weights + convert to bf16.
//   GkT/UkT: [H, D] (row h = column h of Gk/Uk, K=D contiguous)
//   DkT    : [D, H] (row n = column n of Dk, K=H contiguous)
// ---------------------------------------------------------------------------
__global__ __launch_bounds__(256) void k_prep(const float* __restrict__ gk,
                                              const float* __restrict__ uk,
                                              const float* __restrict__ dk,
                                              __bf16* __restrict__ GkT,
                                              __bf16* __restrict__ UkT,
                                              __bf16* __restrict__ DkT) {
    int i = blockIdx.x * 256 + threadIdx.x;          // 0 .. H*D-1 (1,048,576)
    int h = i >> 9, d = i & (D_DIM - 1);             // i = h*512 + d
    GkT[i] = (__bf16)gk[(size_t)d * H_DIM + h];
    UkT[i] = (__bf16)uk[(size_t)d * H_DIM + h];
    int n = i >> 11, k = i & (H_DIM - 1);            // i = n*2048 + k
    DkT[i] = (__bf16)dk[(size_t)k * D_DIM + n];
}

// ---------------------------------------------------------------------------
// Kernel 1: router + alphas + Z = bf16(x*scale). One wave32 per token.
// ---------------------------------------------------------------------------
__global__ __launch_bounds__(256) void k_router(const float* __restrict__ x,
                                                const float* __restrict__ rw,
                                                const float* __restrict__ scale,
                                                __bf16* __restrict__ Z,
                                                float* __restrict__ alpha0,
                                                float* __restrict__ alpha1,
                                                float* __restrict__ usage) {
    __shared__ float s_rwT[E_DIM * D_DIM];   // transposed: [e][d], conflict-free reads
    __shared__ float s_scale[D_DIM];

    int tid = threadIdx.x;
    for (int i = tid; i < E_DIM * D_DIM; i += 256) {
        int e = i >> 9, d = i & (D_DIM - 1);
        s_rwT[i] = rw[d * E_DIM + e];
    }
    for (int i = tid; i < D_DIM; i += 256) s_scale[i] = scale[i];
    __syncthreads();

    int wave = tid >> 5, lane = tid & 31;
    int token = blockIdx.x * 8 + wave;
    const float* xt = x + (size_t)token * D_DIM;
    __bf16* zt = Z + (size_t)token * D_DIM;

    float ss = 0.0f;
    float lg[E_DIM] = {0.f, 0.f, 0.f, 0.f, 0.f, 0.f, 0.f, 0.f};
#pragma unroll
    for (int k = 0; k < 16; ++k) {
        int d = lane + 32 * k;
        float xv = xt[d];
        ss += xv * xv;
#pragma unroll
        for (int e = 0; e < E_DIM; ++e) lg[e] += xv * s_rwT[e * D_DIM + d];
        zt[d] = (__bf16)(xv * s_scale[d]);
    }
    // wave32 reduction
#pragma unroll
    for (int off = 16; off > 0; off >>= 1) {
        ss += __shfl_xor(ss, off, 32);
#pragma unroll
        for (int e = 0; e < E_DIM; ++e) lg[e] += __shfl_xor(lg[e], off, 32);
    }

    if (lane == 0) {
        float var = ss * (1.0f / (float)D_DIM);
        float mx = lg[0];
#pragma unroll
        for (int e = 1; e < E_DIM; ++e) mx = fmaxf(mx, lg[e]);
        float p[E_DIM], sum = 0.0f;
#pragma unroll
        for (int e = 0; e < E_DIM; ++e) { p[e] = __expf(lg[e] - mx); sum += p[e]; }
        float inv = 1.0f / sum;
#pragma unroll
        for (int e = 0; e < E_DIM; ++e) p[e] *= inv;
        // top-2 (order irrelevant: output is symmetric sum)
        int a = 0;
#pragma unroll
        for (int e = 1; e < E_DIM; ++e) if (p[e] > p[a]) a = e;
        int b = (a == 0) ? 1 : 0;
#pragma unroll
        for (int e = 0; e < E_DIM; ++e) if (e != a && p[e] > p[b]) b = e;
        float s2 = p[a] + p[b];
        float w0 = p[a] / s2, w1 = p[b] / s2;
        alpha0[token] = w0 * rsqrtf(w0 * w0 * var + 1e-6f);
        alpha1[token] = w1 * rsqrtf(w1 * w1 * var + 1e-6f);
#pragma unroll
        for (int e = 0; e < E_DIM; ++e) atomicAdd(&usage[e], p[e]);
    }
}

// ---------------------------------------------------------------------------
// Kernel 2: fused gate+up GEMM (Z[BS,512] x GkT/UkT -> 128x64 tiles) with
// per-expert silu combine epilogue -> P bf16 [BS, 2048].
// WG: 256 threads = 8 waves (4 M x 2 N), wave tile 32x32, WMMA 16x16x32 bf16.
// Double-buffered LDS fed by global_load_async_to_lds_b128 (ASYNCcnt).
// ---------------------------------------------------------------------------
__global__ __launch_bounds__(256) void k_gateup(const __bf16* __restrict__ Z,
                                                const __bf16* __restrict__ GkT,
                                                const __bf16* __restrict__ UkT,
                                                const float* __restrict__ gb,
                                                const float* __restrict__ ub,
                                                const float* __restrict__ alpha0,
                                                const float* __restrict__ alpha1,
                                                __bf16* __restrict__ P) {
    __shared__ alignas(16) __bf16 lA[2][128 * LDA];
    __shared__ alignas(16) __bf16 lG[2][64 * LDA];
    __shared__ alignas(16) __bf16 lU[2][64 * LDA];

    const int tid = threadIdx.x;
    const int wave = tid >> 5, lane = tid & 31;
    const int wm = (wave >> 1) * 32;       // wave M offset within 128
    const int wn = (wave & 1) * 32;        // wave N offset within 64
    const int tok0 = blockIdx.x * 128;
    const int h0 = blockIdx.y * 64;

    // per-thread staging chunk: row 0..63, 8-half part within 32-wide K slice
    const int row = tid >> 2;
    const int part = (tid & 3) * 8;
    const unsigned offAB = (unsigned)((row * LDA + part) * 2);  // bytes, 16B aligned
    const unsigned ldsA1 = (unsigned)(uintptr_t)(&lA[0][0]) + offAB;
    const unsigned ldsA2 = ldsA1 + (unsigned)(64 * LDA * 2);
    const unsigned ldsG  = (unsigned)(uintptr_t)(&lG[0][0]) + offAB;
    const unsigned ldsU  = (unsigned)(uintptr_t)(&lU[0][0]) + offAB;
    const unsigned strA = 128 * LDA * 2, strB = 64 * LDA * 2;

    const __bf16* gA1 = &Z[(size_t)(tok0 + row) * D_DIM + part];
    const __bf16* gA2 = &Z[(size_t)(tok0 + row + 64) * D_DIM + part];
    const __bf16* gG  = &GkT[(size_t)(h0 + row) * D_DIM + part];
    const __bf16* gU  = &UkT[(size_t)(h0 + row) * D_DIM + part];

    auto issue = [&](int buf, int k0) {
        async_b128(ldsA1 + buf * strA, (unsigned long long)(uintptr_t)(gA1 + k0));
        async_b128(ldsA2 + buf * strA, (unsigned long long)(uintptr_t)(gA2 + k0));
        async_b128(ldsG  + buf * strB, (unsigned long long)(uintptr_t)(gG  + k0));
        async_b128(ldsU  + buf * strB, (unsigned long long)(uintptr_t)(gU  + k0));
    };

    const floatx8 zero8 = {0.f, 0.f, 0.f, 0.f, 0.f, 0.f, 0.f, 0.f};
    floatx8 accG[2][2], accU[2][2];
#pragma unroll
    for (int i = 0; i < 2; ++i)
#pragma unroll
        for (int j = 0; j < 2; ++j) { accG[i][j] = zero8; accU[i][j] = zero8; }

    const int lr = lane & 15;
    const int kb8 = (lane >> 4) * 8;

    issue(0, 0);
    const int NK = D_DIM / 32;
    for (int kb = 0; kb < NK; ++kb) {
        const int cur = kb & 1;
        wait_async0();            // my DMA into buf[cur] done
        __syncthreads();          // everyone's DMA done; prior reads of buf[cur^1] done
        if (kb + 1 < NK) issue(cur ^ 1, (kb + 1) * 32);

        const __bf16* A = lA[cur];
        const __bf16* G = lG[cur];
        const __bf16* U = lU[cur];
        bf16x16 af[2], gf[2], uf[2];
#pragma unroll
        for (int mt = 0; mt < 2; ++mt) af[mt] = load_frag(&A[(wm + mt * 16 + lr) * LDA], kb8);
#pragma unroll
        for (int nt = 0; nt < 2; ++nt) {
            gf[nt] = load_frag(&G[(wn + nt * 16 + lr) * LDA], kb8);
            uf[nt] = load_frag(&U[(wn + nt * 16 + lr) * LDA], kb8);
        }
#pragma unroll
        for (int mt = 0; mt < 2; ++mt)
#pragma unroll
            for (int nt = 0; nt < 2; ++nt) {
                accG[mt][nt] = __builtin_amdgcn_wmma_f32_16x16x32_bf16(
                    false, af[mt], false, gf[nt], (short)0, accG[mt][nt], false, false);
                accU[mt][nt] = __builtin_amdgcn_wmma_f32_16x16x32_bf16(
                    false, af[mt], false, uf[nt], (short)0, accU[mt][nt], false, false);
            }
    }

    // Epilogue: p = silu(a0*m+gb)*(a0*n+ub) + silu(a1*m+gb)*(a1*n+ub)
    const int ro = (lane >> 4) * 8;
#pragma unroll
    for (int mt = 0; mt < 2; ++mt) {
        int tokBase = tok0 + wm + mt * 16 + ro;
        float a0v[8], a1v[8];
#pragma unroll
        for (int r = 0; r < 8; ++r) { a0v[r] = alpha0[tokBase + r]; a1v[r] = alpha1[tokBase + r]; }
#pragma unroll
        for (int nt = 0; nt < 2; ++nt) {
            int h = h0 + wn + nt * 16 + lr;
            float gbv = gb[h], ubv = ub[h];
#pragma unroll
            for (int r = 0; r < 8; ++r) {
                float m = accG[mt][nt][r];
                float n = accU[mt][nt][r];
                float p = silu_f(a0v[r] * m + gbv) * (a0v[r] * n + ubv)
                        + silu_f(a1v[r] * m + gbv) * (a1v[r] * n + ubv);
                P[(size_t)(tokBase + r) * H_DIM + h] = (__bf16)p;
            }
        }
    }
}

// ---------------------------------------------------------------------------
// Kernel 3: down GEMM: out = P[BS,2048] x DkT(rows=n of D, K=H contiguous) + 2*db
// Same double-buffered async-DMA structure.
// ---------------------------------------------------------------------------
__global__ __launch_bounds__(256) void k_down(const __bf16* __restrict__ P,
                                              const __bf16* __restrict__ DkT,
                                              const float* __restrict__ db,
                                              float* __restrict__ out) {
    __shared__ alignas(16) __bf16 lA[2][128 * LDA];
    __shared__ alignas(16) __bf16 lB[2][64 * LDA];

    const int tid = threadIdx.x;
    const int wave = tid >> 5, lane = tid & 31;
    const int wm = (wave >> 1) * 32;
    const int wn = (wave & 1) * 32;
    const int tok0 = blockIdx.x * 128;
    const int n0 = blockIdx.y * 64;

    const int row = tid >> 2;
    const int part = (tid & 3) * 8;
    const unsigned offAB = (unsigned)((row * LDA + part) * 2);
    const unsigned ldsA1 = (unsigned)(uintptr_t)(&lA[0][0]) + offAB;
    const unsigned ldsA2 = ldsA1 + (unsigned)(64 * LDA * 2);
    const unsigned ldsB  = (unsigned)(uintptr_t)(&lB[0][0]) + offAB;
    const unsigned strA = 128 * LDA * 2, strB = 64 * LDA * 2;

    const __bf16* gA1 = &P[(size_t)(tok0 + row) * H_DIM + part];
    const __bf16* gA2 = &P[(size_t)(tok0 + row + 64) * H_DIM + part];
    const __bf16* gB  = &DkT[(size_t)(n0 + row) * H_DIM + part];

    auto issue = [&](int buf, int k0) {
        async_b128(ldsA1 + buf * strA, (unsigned long long)(uintptr_t)(gA1 + k0));
        async_b128(ldsA2 + buf * strA, (unsigned long long)(uintptr_t)(gA2 + k0));
        async_b128(ldsB  + buf * strB, (unsigned long long)(uintptr_t)(gB  + k0));
    };

    const floatx8 zero8 = {0.f, 0.f, 0.f, 0.f, 0.f, 0.f, 0.f, 0.f};
    floatx8 acc[2][2];
#pragma unroll
    for (int i = 0; i < 2; ++i)
#pragma unroll
        for (int j = 0; j < 2; ++j) acc[i][j] = zero8;

    const int lr = lane & 15;
    const int kb8 = (lane >> 4) * 8;

    issue(0, 0);
    const int NK = H_DIM / 32;
    for (int kb = 0; kb < NK; ++kb) {
        const int cur = kb & 1;
        wait_async0();
        __syncthreads();
        if (kb + 1 < NK) issue(cur ^ 1, (kb + 1) * 32);

        const __bf16* A = lA[cur];
        const __bf16* B = lB[cur];
        bf16x16 af[2], bfr[2];
#pragma unroll
        for (int mt = 0; mt < 2; ++mt) af[mt] = load_frag(&A[(wm + mt * 16 + lr) * LDA], kb8);
#pragma unroll
        for (int nt = 0; nt < 2; ++nt) bfr[nt] = load_frag(&B[(wn + nt * 16 + lr) * LDA], kb8);
#pragma unroll
        for (int mt = 0; mt < 2; ++mt)
#pragma unroll
            for (int nt = 0; nt < 2; ++nt)
                acc[mt][nt] = __builtin_amdgcn_wmma_f32_16x16x32_bf16(
                    false, af[mt], false, bfr[nt], (short)0, acc[mt][nt], false, false);
    }

    const int ro = (lane >> 4) * 8;
#pragma unroll
    for (int mt = 0; mt < 2; ++mt) {
        int tokBase = tok0 + wm + mt * 16 + ro;
#pragma unroll
        for (int nt = 0; nt < 2; ++nt) {
            int n = n0 + wn + nt * 16 + lr;
            float dbv = 2.0f * db[n];
#pragma unroll
            for (int r = 0; r < 8; ++r)
                out[(size_t)(tokBase + r) * D_DIM + n] = acc[mt][nt][r] + dbv;
        }
    }
}

// ---------------------------------------------------------------------------
// Kernel 4: load-balancing loss scalar
// ---------------------------------------------------------------------------
__global__ void k_loss(const float* __restrict__ usage, float* __restrict__ out_loss) {
    if (threadIdx.x == 0 && blockIdx.x == 0) {
        const float inv = 1.0f / (float)BS_TOK;
        float loss = 0.0f;
#pragma unroll
        for (int e = 0; e < E_DIM; ++e) {
            float u = usage[e] * inv;
            loss -= u * __logf(u + 1e-6f);
        }
        *out_loss = loss;
    }
}

// ---------------------------------------------------------------------------
// Host launcher
// ---------------------------------------------------------------------------
extern "C" void kernel_launch(void* const* d_in, const int* in_sizes, int n_in,
                              void* d_out, int out_size, void* d_ws, size_t ws_size,
                              hipStream_t stream) {
    const float* x     = (const float*)d_in[0];   // [16,4096,512]
    const float* rw    = (const float*)d_in[1];   // [512,8]
    const float* scale = (const float*)d_in[2];   // [512]
    const float* gk    = (const float*)d_in[3];   // [512,2048]
    const float* gb    = (const float*)d_in[4];   // [2048]
    const float* uk    = (const float*)d_in[5];   // [512,2048]
    const float* ub    = (const float*)d_in[6];   // [2048]
    const float* dk    = (const float*)d_in[7];   // [2048,512]
    const float* db    = (const float*)d_in[8];   // [512]

    float* out = (float*)d_out;                   // [16,4096,512] + 1 loss
    float* out_loss = out + (size_t)BS_TOK * D_DIM;

    char* ws = (char*)d_ws;
    size_t off = 0;
    __bf16* Z      = (__bf16*)(ws + off); off += (size_t)BS_TOK * D_DIM * 2;   // 67,108,864
    __bf16* GkT    = (__bf16*)(ws + off); off += (size_t)H_DIM * D_DIM * 2;    //  2,097,152
    __bf16* UkT    = (__bf16*)(ws + off); off += (size_t)H_DIM * D_DIM * 2;
    __bf16* DkT    = (__bf16*)(ws + off); off += (size_t)D_DIM * H_DIM * 2;
    float*  alpha0 = (float*)(ws + off);  off += (size_t)BS_TOK * 4;
    float*  alpha1 = (float*)(ws + off);  off += (size_t)BS_TOK * 4;
    float*  usage  = (float*)(ws + off);  off += 256;
    __bf16* P      = (__bf16*)(ws + off); off += (size_t)BS_TOK * H_DIM * 2;   // 268,435,456

    k_init<<<1, 32, 0, stream>>>(usage);
    k_prep<<<(H_DIM * D_DIM) / 256, 256, 0, stream>>>(gk, uk, dk, GkT, UkT, DkT);
    k_router<<<BS_TOK / 8, 256, 0, stream>>>(x, rw, scale, Z, alpha0, alpha1, usage);
    k_gateup<<<dim3(BS_TOK / 128, H_DIM / 64), 256, 0, stream>>>(Z, GkT, UkT, gb, ub,
                                                                 alpha0, alpha1, P);
    k_down<<<dim3(BS_TOK / 128, D_DIM / 64), 256, 0, stream>>>(P, DkT, db, out);
    k_loss<<<1, 32, 0, stream>>>(usage, out_loss);
}
